// ContrastiveLearningLoss_6416681140574
// MI455X (gfx1250) — matrix operand order repaired and checked by reading, not compiled, not run on hardware
//
#include <hip/hip_runtime.h>
#include <math.h>

// ---------------- problem constants (from setup_inputs) ----------------
#define BB     8
#define LL     1024
#define DIN    512
#define DF     256
#define NN     (BB * LL)          // 8192
#define CSPLIT 8                  // column chunks for the fused sim+LSE kernel
#define CTILES ((NN / CSPLIT) / 16)  // 64 column tiles per chunk

typedef __attribute__((ext_vector_type(16))) __bf16 v16bf;
typedef __attribute__((ext_vector_type(8)))  __bf16 v8bf;
typedef __attribute__((ext_vector_type(8)))  float  v8f;
typedef __attribute__((ext_vector_type(2)))  float  v2f;
typedef __attribute__((ext_vector_type(4)))  unsigned int u32x4;
typedef __attribute__((ext_vector_type(8)))  int i32x8;
typedef __attribute__((ext_vector_type(4)))  int i32x4;

union bf16x16u { v16bf v; v8bf h[2]; };

static __device__ __forceinline__ unsigned short f32_to_bf16(float f) {
  unsigned int u = __float_as_uint(f);
  unsigned int rnd = 0x7FFFu + ((u >> 16) & 1u);   // round-to-nearest-even
  u += rnd;
  return (unsigned short)(u >> 16);
}

// ---------------------------------------------------------------------
// TDM: issue a 2D tile load (tile_d0 elems x tile_d1 rows, 2-byte elems)
// from global to LDS. Descriptor per CDNA5 D# layout (CDNA5 ISA).
// ---------------------------------------------------------------------
static __device__ __forceinline__ void tdm_load_tile_2d(
    unsigned lds_byte_addr, const void* gaddr,
    unsigned tile_d0, unsigned tile_d1,
    unsigned tensor_d0, unsigned tensor_d1, unsigned stride_d0) {
  unsigned long long ga = (unsigned long long)(uintptr_t)gaddr;
  u32x4 g0;
  g0[0] = 1u;                                        // count=1 (valid), user mode
  g0[1] = lds_byte_addr;                             // lds_addr
  g0[2] = (unsigned)(ga & 0xFFFFFFFFu);              // global_addr[31:0]
  g0[3] = (unsigned)((ga >> 32) & 0x01FFFFFFu)       // global_addr[56:32]
        | (2u << 30);                                // type = 2 ("image")
  i32x8 g1;
  g1[0] = (int)(1u << 16);                           // data_size=1 (2 bytes)
  g1[1] = (int)(tensor_d0 << 16);                    // tensor_dim0[15:0]
  g1[2] = (int)((tensor_d0 >> 16) | (tensor_d1 << 16));
  g1[3] = (int)((tensor_d1 >> 16) | (tile_d0 << 16));// tile_dim0
  g1[4] = (int)(tile_d1 & 0xFFFFu);                  // tile_dim1 (tile_dim2=0)
  g1[5] = (int)stride_d0;                            // tensor_dim0_stride[31:0]
  g1[6] = 0;                                         // stride hi + dim1_stride lo
  g1[7] = 0;
  i32x4 z4 = {0, 0, 0, 0};
#if __clang_major__ >= 23
  i32x8 z8 = {0, 0, 0, 0, 0, 0, 0, 0};
  __builtin_amdgcn_tensor_load_to_lds(g0, g1, z4, z4, z8, 0);
#else
  __builtin_amdgcn_tensor_load_to_lds(g0, g1, z4, z4, 0);
#endif
}

// ---------------------------------------------------------------------
// Kernel 1: emb = relu(X @ W + b), native fp32 WMMA (16x16x4_f32).
// Each wave computes one 16x16 output tile; block = 8 waves = 16 rows x 128 cols.
// Writes both fp32 (for exact positives) and bf16 (for the sim GEMM).
// ---------------------------------------------------------------------
__global__ void proj_kernel(const float* __restrict__ X1, const float* __restrict__ X2,
                            const float* __restrict__ W,  const float* __restrict__ bias,
                            float* __restrict__ e1f, float* __restrict__ e2f,
                            unsigned short* __restrict__ e1b, unsigned short* __restrict__ e2b) {
  const int lane = threadIdx.x & 31;
  const int wave = threadIdx.x >> 5;
  const int lo = lane & 15;
  const int hi = lane >> 4;

  const int rowBase = blockIdx.x * 16;
  const int colBase = blockIdx.y * 128 + wave * 16;
  const float* X          = blockIdx.z ? X2 : X1;
  float* ef               = blockIdx.z ? e2f : e1f;
  unsigned short* eb      = blockIdx.z ? e2b : e1b;

  const float* xrow = X + (size_t)(rowBase + lo) * DIN;
  const int col = colBase + lo;

  v8f c = {};
  for (int k = 0; k < DIN; k += 4) {
    // A 16x4 f32: lanes 0-15 K={0,1}, lanes 16-31 K={2,3} (contiguous pair)
    v2f a = *(const v2f*)(xrow + k + hi * 2);
    // B 4x16 f32: lane column = lo, same K pair, stride DF between K rows
    v2f b;
    b[0] = W[(size_t)(k + hi * 2 + 0) * DF + col];
    b[1] = W[(size_t)(k + hi * 2 + 1) * DF + col];
    c = __builtin_amdgcn_wmma_f32_16x16x4_f32(false, a, false, b, (short)0, c, false, false);
  }

  const float bv = bias[col];
#pragma unroll
  for (int r = 0; r < 8; ++r) {
    const int row = rowBase + r + 8 * hi;          // C/D layout: VGPR r -> M=r (+8 for hi half)
    float v = c[r] + bv;
    v = v > 0.0f ? v : 0.0f;
    ef[(size_t)row * DF + col] = v;
    eb[(size_t)row * DF + col] = f32_to_bf16(v);
  }
}

// ---------------------------------------------------------------------
// Kernel 2: exact fp32 positives. One wave per row n = (b,l):
//   pos = e1.e2 + mean_band(e1,e1,rs) + mean_band(e2,e2,rs) + mean_band(e1,e2,rt)
// ---------------------------------------------------------------------
__global__ void pos_kernel(const float* __restrict__ e1f, const float* __restrict__ e2f,
                           const int* __restrict__ prs, const int* __restrict__ prt,
                           float* __restrict__ pos) {
  const int lane = threadIdx.x & 31;
  const int wave = threadIdx.x >> 5;
  const int n = blockIdx.x * 8 + wave;
  const int l = n & (LL - 1);
  const int d0 = lane * (DF / 32);                 // 8 dims per lane
  const float* a1 = e1f + (size_t)n * DF;
  const float* a2 = e2f + (size_t)n * DF;

  float acc = 0.0f;
#pragma unroll
  for (int d = 0; d < DF / 32; ++d) acc += a1[d0 + d] * a2[d0 + d];

  const int rs = prs[0];
  const int rt = prt[0];

  if (rs > 0) {
    const int j0 = (l - rs < 0) ? 0 : (l - rs);
    const int j1 = (l + rs + 1 > LL) ? LL : (l + rs + 1);
    const float inv = 1.0f / (float)(j1 - j0);
    float band = 0.0f;
    for (int j = j0; j < j1; ++j) {
      const float* b1 = e1f + (size_t)(n - l + j) * DF;
      const float* b2 = e2f + (size_t)(n - l + j) * DF;
#pragma unroll
      for (int d = 0; d < DF / 32; ++d)
        band += a1[d0 + d] * b1[d0 + d] + a2[d0 + d] * b2[d0 + d];
    }
    acc += band * inv;
  }
  if (rt > 0) {
    const int j0 = (l - rt < 0) ? 0 : (l - rt);
    const int j1 = (l + rt + 1 > LL) ? LL : (l + rt + 1);
    const float inv = 1.0f / (float)(j1 - j0);
    float band = 0.0f;
    for (int j = j0; j < j1; ++j) {
      const float* b2 = e2f + (size_t)(n - l + j) * DF;
#pragma unroll
      for (int d = 0; d < DF / 32; ++d) band += a1[d0 + d] * b2[d0 + d];
    }
    acc += band * inv;
  }

  for (int off = 16; off > 0; off >>= 1) acc += __shfl_xor(acc, off, 32);
  if (lane == 0) pos[n] = acc;
}

// ---------------------------------------------------------------------
// Kernel 3: fused sim GEMM (bf16 WMMA, fp32 accum) + online logsumexp.
// Block = 8 row strips sharing one 1024-column chunk. Each e2 tile
// (16 rows x 256 K bf16 = 8KB) is staged into LDS once per block by the
// Tensor Data Mover (double-buffered, TENSORcnt), then all 8 waves feed
// their v_wmma_f32_16x16x32_bf16 from LDS. A strips stay in VGPRs.
// ---------------------------------------------------------------------
__global__ void sim_lse_kernel(const unsigned short* __restrict__ e1b,
                               const unsigned short* __restrict__ e2b,
                               float* __restrict__ pm, float* __restrict__ ps) {
  __shared__ unsigned short Bs[2][16 * DF];        // 2 x 8KB double buffer

  const int lane = threadIdx.x & 31;
  const int wave = threadIdx.x >> 5;
  const int chunk = blockIdx.x & (CSPLIT - 1);     // 0..7
  const int stripGroup = blockIdx.x >> 3;          // 0..63
  const int strip = stripGroup * 8 + wave;         // 0..511
  const int rowBase = strip * 16;
  const int colBase = chunk * (NN / CSPLIT);
  const int lo = lane & 15;
  const int hi = lane >> 4;

  // A 16x32 bf16 layout: lane row = lo; lanes 0-15 hold K runs {0..7,16..23},
  // lanes 16-31 hold {8..15,24..31} -> two contiguous 16B chunks per k-step.
  v16bf A[8];
  const unsigned short* arow = e1b + (size_t)(rowBase + lo) * DF;
#pragma unroll
  for (int kk = 0; kk < 8; ++kk) {
    bf16x16u u;
    u.h[0] = *(const v8bf*)(arow + kk * 32 + hi * 8);
    u.h[1] = *(const v8bf*)(arow + kk * 32 + 16 + hi * 8);
    A[kk] = u.v;
  }

  float m[8], s[8];
#pragma unroll
  for (int r = 0; r < 8; ++r) { m[r] = -3.0e38f; s[r] = 0.0f; }

  // prologue: TDM-load tile 0 into buffer 0 (wave 0 issues for the block)
  if (wave == 0) {
    tdm_load_tile_2d((unsigned)(uintptr_t)&Bs[0][0],
                     e2b + (size_t)colBase * DF,
                     DF, 16, DF, NN, DF);
  }

  for (int t = 0; t < CTILES; ++t) {
    if (wave == 0) {
      if (t + 1 < CTILES) {
        tdm_load_tile_2d((unsigned)(uintptr_t)&Bs[(t + 1) & 1][0],
                         e2b + (size_t)(colBase + (t + 1) * 16) * DF,
                         DF, 16, DF, NN, DF);
        __builtin_amdgcn_s_wait_tensorcnt(1);      // tile t landed (in-order)
      } else {
        __builtin_amdgcn_s_wait_tensorcnt(0);
      }
    }
    __syncthreads();                               // LDS tile t visible to all waves

    // B 32x16 bf16 layout: lane column = lo; lanes 0-15 K {0..7,16..23}, lanes 16-31
    // K {8..15,24..31} -> two contiguous 16B runs of an e2 row per lane per k-step.
    const unsigned short* brow = &Bs[t & 1][lo * DF];
    v8f c = {};
#pragma unroll
    for (int kk = 0; kk < 8; ++kk) {
      bf16x16u u;
      u.h[0] = *(const v8bf*)(brow + kk * 32 + hi * 8);
      u.h[1] = *(const v8bf*)(brow + kk * 32 + 16 + hi * 8);
      c = __builtin_amdgcn_wmma_f32_16x16x32_bf16(false, A[kk], false, u.v, (short)0, c, false, false);
    }
#pragma unroll
    for (int r = 0; r < 8; ++r) {                  // per-lane online logsumexp
      const float x  = c[r];
      const float mo = m[r];
      const float mn = fmaxf(mo, x);
      s[r] = s[r] * __expf(mo - mn) + __expf(x - mn);
      m[r] = mn;
    }
    __syncthreads();                               // done reading buf before reuse
  }

  // merge (m,s) across the 16 lanes of each half-wave (row lives in one half)
  for (int off = 1; off < 16; off <<= 1) {
#pragma unroll
    for (int r = 0; r < 8; ++r) {
      const float m2 = __shfl_xor(m[r], off, 32);
      const float s2 = __shfl_xor(s[r], off, 32);
      const float mn = fmaxf(m[r], m2);
      s[r] = s[r] * __expf(m[r] - mn) + s2 * __expf(m2 - mn);
      m[r] = mn;
    }
  }

  if (lo == 0) {
#pragma unroll
    for (int r = 0; r < 8; ++r) {
      const int row = rowBase + r + 8 * hi;
      pm[(size_t)row * CSPLIT + chunk] = m[r];
      ps[(size_t)row * CSPLIT + chunk] = s[r];
    }
  }
}

// ---------------------------------------------------------------------
// Kernel 4: merge chunk partials -> per-row LSE, combine with positives,
// mean over N. Single block, no atomics (d_out is poisoned by the harness).
// ---------------------------------------------------------------------
__global__ void reduce_kernel(const float* __restrict__ pos,
                              const float* __restrict__ pm, const float* __restrict__ ps,
                              float* __restrict__ out) {
  __shared__ float sm[1024];
  const int t = threadIdx.x;
  const float logN = logf((float)NN);
  float acc = 0.0f;
  for (int row = t; row < NN; row += 1024) {
    float m = pm[(size_t)row * CSPLIT];
    float s = ps[(size_t)row * CSPLIT];
    for (int p = 1; p < CSPLIT; ++p) {
      const float m2 = pm[(size_t)row * CSPLIT + p];
      const float s2 = ps[(size_t)row * CSPLIT + p];
      const float mn = fmaxf(m, m2);
      s = s * __expf(m - mn) + s2 * __expf(m2 - mn);
      m = mn;
    }
    const float negative = m + logf(s) - logN;
    acc += negative - pos[row];
  }
  sm[t] = acc;
  __syncthreads();
  for (int stride = 512; stride > 0; stride >>= 1) {
    if (t < stride) sm[t] += sm[t + stride];
    __syncthreads();
  }
  if (t == 0) out[0] = sm[0] / (float)NN;
}

// ---------------------------------------------------------------------
extern "C" void kernel_launch(void* const* d_in, const int* in_sizes, int n_in,
                              void* d_out, int out_size, void* d_ws, size_t ws_size,
                              hipStream_t stream) {
  (void)in_sizes; (void)n_in; (void)out_size; (void)ws_size;

  const float* feature1 = (const float*)d_in[0];
  const float* feature2 = (const float*)d_in[1];
  const float* W        = (const float*)d_in[2];
  const float* b        = (const float*)d_in[3];
  const int*   prs      = (const int*)d_in[4];
  const int*   prt      = (const int*)d_in[5];
  float* out            = (float*)d_out;

  // workspace carve-up (all 256B aligned)
  char* ws = (char*)d_ws;
  size_t off = 0;
  float* e1f = (float*)(ws + off);           off += (size_t)NN * DF * sizeof(float);   // 8 MB
  float* e2f = (float*)(ws + off);           off += (size_t)NN * DF * sizeof(float);   // 8 MB
  unsigned short* e1b = (unsigned short*)(ws + off); off += (size_t)NN * DF * 2;       // 4 MB
  unsigned short* e2b = (unsigned short*)(ws + off); off += (size_t)NN * DF * 2;       // 4 MB
  float* pos = (float*)(ws + off);           off += (size_t)NN * sizeof(float);
  float* pm  = (float*)(ws + off);           off += (size_t)NN * CSPLIT * sizeof(float);
  float* ps  = (float*)(ws + off);           off += (size_t)NN * CSPLIT * sizeof(float);

  // 1) projection: grid = (row tiles, col super-tiles, feature), 8 waves/block
  proj_kernel<<<dim3(NN / 16, DF / 128, 2), 256, 0, stream>>>(
      feature1, feature2, W, b, e1f, e2f, e1b, e2b);

  // 2) positives: one wave per row
  pos_kernel<<<dim3(NN / 8), 256, 0, stream>>>(e1f, e2f, prs, prt, pos);

  // 3) fused sim GEMM (TDM-staged B tiles) + online logsumexp
  sim_lse_kernel<<<dim3((NN / 16) * CSPLIT / 8), 256, 0, stream>>>(e1b, e2b, pm, ps);

  // 4) final merge + mean
  reduce_kernel<<<dim3(1), 1024, 0, stream>>>(pos, pm, ps, out);
}
